// MultiHeadAttention_4294967296395
// MI455X (gfx1250) — matrix operand order repaired and checked by reading, not compiled, not run on hardware
//
#include <hip/hip_runtime.h>
#include <hip/hip_bf16.h>
#include <stdint.h>

#define EMBED 1024
#define HEADS 16
#define HEADD 64
#define BATCH 8
#define SEQ   1024

typedef __attribute__((ext_vector_type(16))) __bf16 v16bf;
typedef __attribute__((ext_vector_type(8)))  float  v8f;

union BF16Frag {
    v16bf v;
    uint4 q[2];
    __bf16 e[16];
};

static __device__ __forceinline__ v8f wmma_bf16(const v16bf& a, const v16bf& b, const v8f& c) {
    // D = A(16x32 bf16) x B(32x16 bf16) + C(16x16 f32)
    return __builtin_amdgcn_wmma_f32_16x16x32_bf16(false, a, false, b, (short)0, c, false, false);
}

// ---------------------------------------------------------------------------
// Async global->LDS staging (CDNA5 GLOBAL_LOAD_ASYNC_TO_LDS_B128, ASYNCcnt).
// Builtin signature (from compiler diagnostic): 
//   (int4 addrspace(1)* gsrc, int4 addrspace(3)* ldst, imm offset, imm cpol)
// ---------------------------------------------------------------------------
#if __has_builtin(__builtin_amdgcn_global_load_async_to_lds_b128) && \
    __has_builtin(__builtin_amdgcn_s_wait_asynccnt)
#define HAVE_ASYNC_LDS 1
#else
#define HAVE_ASYNC_LDS 0
#endif

typedef int int4v __attribute__((vector_size(16)));
typedef __attribute__((address_space(1))) int4v gint4;
typedef __attribute__((address_space(3))) int4v lint4;

static __device__ __forceinline__ void cp16_async(void* lds, const void* g) {
#if HAVE_ASYNC_LDS
    __builtin_amdgcn_global_load_async_to_lds_b128(
        (gint4*)(uintptr_t)g, (lint4*)(unsigned)(uintptr_t)lds, 0, 0);
#else
    *(uint4*)lds = *(const uint4*)g;
#endif
}

static __device__ __forceinline__ void wait_async() {
#if HAVE_ASYNC_LDS
    __builtin_amdgcn_s_wait_asynccnt(0);
#endif
}

// ---------------------------------------------------------------------------
// Kernel 1: per-head QKV projections (x @ W.T + b), f32 in -> bf16 planes out.
// Plane layout: P[(b*HEADS+h)*SEQ + s][64], contiguous per head.
// ---------------------------------------------------------------------------
__global__ __launch_bounds__(128)
void proj_qkv_kernel(const float* __restrict__ values,
                     const float* __restrict__ keys,
                     const float* __restrict__ query,
                     const float* __restrict__ Wv, const float* __restrict__ bv,
                     const float* __restrict__ Wk, const float* __restrict__ bk,
                     const float* __restrict__ Wq, const float* __restrict__ bq,
                     __bf16* __restrict__ Qp, __bf16* __restrict__ Kp,
                     __bf16* __restrict__ Vp)
{
    const int tid  = threadIdx.x;
    const int lane = tid & 31;
    const int wave = tid >> 5;
    const int lh   = lane >> 4;   // lane half (K-split in A/B fragments)
    const int l16  = lane & 15;

    const int bh = blockIdx.x;            // b*HEADS + h
    const int b  = bh >> 4;
    const int h  = bh & 15;
    const int sbase = blockIdx.y * 64 + wave * 16;

    const float* Xs[3] = { query, keys, values };
    const float* Ws[3] = { Wq, Wk, Wv };
    const float* Bs[3] = { bq, bk, bv };
    __bf16*      Os[3] = { Qp, Kp, Vp };

    #pragma unroll
    for (int p = 0; p < 3; ++p) {
        // --- A fragment: 16 token rows x 64 head dims (two K=32 chunks) ---
        const int srow = sbase + l16;     // both lane halves load the same row
        const float* xrow = Xs[p] + (size_t)(b * SEQ + srow) * EMBED + h * HEADD;
        BF16Frag a[2];
        #pragma unroll
        for (int kc = 0; kc < 2; ++kc) {
            const int c0 = kc * 32 + lh * 8;        // elems 0..7 :  K = c0..c0+7
            const int c1 = kc * 32 + 16 + lh * 8;   // elems 8..15:  K = c1..c1+7
            #pragma unroll
            for (int i = 0; i < 8; ++i) a[kc].e[i]     = (__bf16)xrow[c0 + i];
            #pragma unroll
            for (int i = 0; i < 8; ++i) a[kc].e[8 + i] = (__bf16)xrow[c1 + i];
        }
        const float* W = Ws[p];
        __bf16* outp = Os[p] + (size_t)bh * SEQ * HEADD;
        #pragma unroll
        for (int nt = 0; nt < 4; ++nt) {
            const int n = nt * 16 + l16;  // output dim (B-fragment column)
            v8f acc = {};
            #pragma unroll
            for (int kc = 0; kc < 2; ++kc) {
                // B[z,n] = W[n,z]  (x @ W.T) : two runs of 8 f32 of W row n, 16 apart
                BF16Frag bfr;
                const float* wrow = W + n * HEADD + kc * 32 + lh * 8;
                #pragma unroll
                for (int i = 0; i < 8; ++i) bfr.e[i] = (__bf16)wrow[i];
                #pragma unroll
                for (int i = 0; i < 8; ++i) bfr.e[8 + i] = (__bf16)wrow[16 + i];
                acc = wmma_bf16(a[kc].v, bfr.v, acc);
            }
            const float bias = Bs[p][n];
            #pragma unroll
            for (int r = 0; r < 8; ++r) {
                const int m = r + 8 * lh;  // C layout: M = vgpr + 8*half
                outp[(size_t)(sbase + m) * HEADD + n] = (__bf16)(acc[r] + bias);
            }
        }
    }
}

// ---------------------------------------------------------------------------
// Kernel 2: causal flash attention per (b,h, 64 q-rows). 4 waves, 16 rows each.
// Double-buffered async K/V staging overlapped with WMMA compute.
// ---------------------------------------------------------------------------
__global__ __launch_bounds__(128)
void flash_attn_kernel(const __bf16* __restrict__ Qp,
                       const __bf16* __restrict__ Kp,
                       const __bf16* __restrict__ Vp,
                       __bf16* __restrict__ AttnOut)
{
    constexpr int LD = 72;                     // padded LDS row stride (bf16)
    __shared__ __bf16 Kt2[2][64 * LD];         // [key][dim], double buffered
    __shared__ __bf16 Vr2[2][64 * LD];         // [key][dim], double buffered
    __shared__ __bf16 Vt[64 * LD];             // transposed [dim][key]
    __shared__ __bf16 Pb[4 * 16 * LD];         // per-wave P staging (C->A)

    const int tid  = threadIdx.x;
    const int lane = tid & 31;
    const int wave = tid >> 5;
    const int lh   = lane >> 4;
    const int l16  = lane & 15;

    const int bh    = blockIdx.x;
    const int b     = bh >> 4;
    const int h     = bh & 15;
    const int qbase = blockIdx.y * 64;
    const int qrow0 = qbase + wave * 16;

    const __bf16* qplane = Qp + (size_t)bh * SEQ * HEADD;
    const __bf16* kplane = Kp + (size_t)bh * SEQ * HEADD;
    const __bf16* vplane = Vp + (size_t)bh * SEQ * HEADD;

    // staging geometry: each thread copies 64 B of one K row and one V row
    const int srow  = tid >> 1;
    const int scol0 = (tid & 1) * 32;

    // Q fragments for this wave's 16 rows (two K=32 chunks over 64 dims)
    BF16Frag qf[2];
    {
        const __bf16* qr = qplane + (size_t)(qrow0 + l16) * HEADD;
        #pragma unroll
        for (int kc = 0; kc < 2; ++kc) {
            qf[kc].q[0] = *(const uint4*)(qr + kc * 32 + lh * 8);
            qf[kc].q[1] = *(const uint4*)(qr + kc * 32 + 16 + lh * 8);
        }
    }

    const v8f zero8 = {};
    v8f acc[4];
    #pragma unroll
    for (int t = 0; t < 4; ++t) acc[t] = zero8;
    float mrun[8], lrun[8];
    #pragma unroll
    for (int r = 0; r < 8; ++r) { mrun[r] = -1e30f; lrun[r] = 0.f; }

    const float scale = 0.03125f;  // 1/sqrt(q_len) = 1/sqrt(1024)

    // prologue: stage tile 0 into buffer 0
    {
        const __bf16* ks = kplane + (size_t)srow * HEADD + scol0;
        const __bf16* vs = vplane + (size_t)srow * HEADD + scol0;
        #pragma unroll
        for (int i = 0; i < 4; ++i) cp16_async(&Kt2[0][srow * LD + scol0 + i * 8], ks + i * 8);
        #pragma unroll
        for (int i = 0; i < 4; ++i) cp16_async(&Vr2[0][srow * LD + scol0 + i * 8], vs + i * 8);
    }

    int it = 0;
    for (int j0 = 0; j0 <= qbase; j0 += 64, ++it) {
        const int cur = it & 1;
        wait_async();            // this wave's async copies for tile `it` done
        __syncthreads();         // all waves' copies visible

        // prefetch next tile into the other buffer (last read of that buffer
        // was before this barrier -> no WAR hazard)
        if (j0 + 64 <= qbase) {
            const __bf16* ks = kplane + (size_t)(j0 + 64 + srow) * HEADD + scol0;
            const __bf16* vs = vplane + (size_t)(j0 + 64 + srow) * HEADD + scol0;
            #pragma unroll
            for (int i = 0; i < 4; ++i)
                cp16_async(&Kt2[cur ^ 1][srow * LD + scol0 + i * 8], ks + i * 8);
            #pragma unroll
            for (int i = 0; i < 4; ++i)
                cp16_async(&Vr2[cur ^ 1][srow * LD + scol0 + i * 8], vs + i * 8);
        }

        // build transposed V tile: Vt[dim][key] from Vr2[cur][key][dim]
        #pragma unroll
        for (int i = 0; i < 32; ++i)
            Vt[(scol0 + i) * LD + srow] = Vr2[cur][srow * LD + scol0 + i];
        __syncthreads();         // Vt ready

        const __bf16* Kt = Kt2[cur];

        // ---- S = Q K^T : 4 key sub-tiles of 16, two K=32 chunks each ----
        v8f s[4];
        #pragma unroll
        for (int nt = 0; nt < 4; ++nt) {
            v8f c = {};
            #pragma unroll
            for (int kc = 0; kc < 2; ++kc) {
                BF16Frag bfr;   // B[z,n] = K[j0+nt*16+n][z]
                const __bf16* kr = &Kt[(nt * 16 + l16) * LD + kc * 32 + lh * 8];
                bfr.q[0] = *(const uint4*)(kr);
                bfr.q[1] = *(const uint4*)(kr + 16);
                c = wmma_bf16(qf[kc].v, bfr.v, c);
            }
            s[nt] = c;
        }

        // scale + analytic causal mask
        #pragma unroll
        for (int nt = 0; nt < 4; ++nt) {
            const int kcol = j0 + nt * 16 + l16;
            #pragma unroll
            for (int r = 0; r < 8; ++r) {
                const int qrow = qrow0 + r + 8 * lh;
                float sv = s[nt][r] * scale;
                if (kcol > qrow) sv = -1e30f;
                s[nt][r] = sv;
            }
        }

        // ---- online softmax; a 16-wide row lives in one VGPR per half ----
        #pragma unroll
        for (int r = 0; r < 8; ++r) {
            float mx = fmaxf(fmaxf(s[0][r], s[1][r]), fmaxf(s[2][r], s[3][r]));
            #pragma unroll
            for (int d = 8; d >= 1; d >>= 1)
                mx = fmaxf(mx, __shfl_xor(mx, d, 32));
            const float mnew  = fmaxf(mrun[r], mx);
            const float alpha = __expf(mrun[r] - mnew);
            float sum = 0.f;
            #pragma unroll
            for (int nt = 0; nt < 4; ++nt) {
                const float pv = __expf(s[nt][r] - mnew);
                s[nt][r] = pv;
                sum += pv;
            }
            #pragma unroll
            for (int d = 8; d >= 1; d >>= 1)
                sum += __shfl_xor(sum, d, 32);
            lrun[r] = lrun[r] * alpha + sum;
            mrun[r] = mnew;
            #pragma unroll
            for (int t = 0; t < 4; ++t) acc[t][r] *= alpha;
        }

        // ---- relayout P (C-frag -> A-frag) via per-wave LDS region ----
        __bf16* pw = &Pb[wave * 16 * LD];
        #pragma unroll
        for (int nt = 0; nt < 4; ++nt)
            #pragma unroll
            for (int r = 0; r < 8; ++r)
                pw[(r + 8 * lh) * LD + nt * 16 + l16] = (__bf16)s[nt][r];

        // ---- O += P V : two K=32 chunks over 64 keys, 4 dim sub-tiles ----
        #pragma unroll
        for (int kc = 0; kc < 2; ++kc) {
            BF16Frag pf;
            const __bf16* pr = &pw[l16 * LD + kc * 32 + lh * 8];
            pf.q[0] = *(const uint4*)(pr);
            pf.q[1] = *(const uint4*)(pr + 16);
            #pragma unroll
            for (int dt = 0; dt < 4; ++dt) {
                BF16Frag vf;    // B[k,d] = V[j0+k][dt*16+l16] (transposed tile)
                const __bf16* vr = &Vt[(dt * 16 + l16) * LD + kc * 32 + lh * 8];
                vf.q[0] = *(const uint4*)(vr);
                vf.q[1] = *(const uint4*)(vr + 16);
                acc[dt] = wmma_bf16(pf.v, vf.v, acc[dt]);
            }
        }
    }

    // normalize by row sums, write bf16 [b][s][h*64+d]
    #pragma unroll
    for (int dt = 0; dt < 4; ++dt)
        #pragma unroll
        for (int r = 0; r < 8; ++r) {
            const int qrow = qrow0 + r + 8 * lh;
            const float o  = acc[dt][r] / lrun[r];
            AttnOut[(size_t)(b * SEQ + qrow) * EMBED + h * HEADD + dt * 16 + l16] =
                (__bf16)o;
        }
}

// ---------------------------------------------------------------------------
// Kernel 3: out = AttnOut @ Wfc.T + bfc  (bf16 GEMM, f32 out), async W staging.
// ---------------------------------------------------------------------------
__global__ __launch_bounds__(128)
void fc_kernel(const __bf16* __restrict__ X,    // [B*S, EMBED] bf16
               const __bf16* __restrict__ Wb,   // [EMBED, EMBED] bf16 (Wfc rows)
               const float* __restrict__ bias,
               float* __restrict__ out)
{
    constexpr int LD = 72;
    __shared__ __bf16 Wt2[2][64 * LD];

    const int tid  = threadIdx.x;
    const int lane = tid & 31;
    const int wave = tid >> 5;
    const int lh   = lane >> 4;
    const int l16  = lane & 15;

    const int trow0 = blockIdx.x * 64 + wave * 16;  // token rows
    const int ncol0 = blockIdx.y * 64;              // output columns

    const int srow  = tid >> 1;          // staging: 64 B of one Wfc row / thread
    const int scol0 = (tid & 1) * 32;

    const v8f zero8 = {};
    v8f acc[4];
    #pragma unroll
    for (int t = 0; t < 4; ++t) acc[t] = zero8;

    // prologue: stage kb=0 into buffer 0
    {
        const __bf16* src = Wb + (size_t)(ncol0 + srow) * EMBED + scol0;
        #pragma unroll
        for (int i = 0; i < 4; ++i)
            cp16_async(&Wt2[0][srow * LD + scol0 + i * 8], src + i * 8);
    }

    int it = 0;
    for (int kb = 0; kb < EMBED; kb += 64, ++it) {
        const int cur = it & 1;
        wait_async();
        __syncthreads();
        if (kb + 64 < EMBED) {
            const __bf16* src = Wb + (size_t)(ncol0 + srow) * EMBED + kb + 64 + scol0;
            #pragma unroll
            for (int i = 0; i < 4; ++i)
                cp16_async(&Wt2[cur ^ 1][srow * LD + scol0 + i * 8], src + i * 8);
        }
        const __bf16* Wt = Wt2[cur];

        const __bf16* xr = X + (size_t)(trow0 + l16) * EMBED + kb;
        #pragma unroll
        for (int kc = 0; kc < 2; ++kc) {
            BF16Frag af;
            af.q[0] = *(const uint4*)(xr + kc * 32 + lh * 8);
            af.q[1] = *(const uint4*)(xr + kc * 32 + 16 + lh * 8);
            #pragma unroll
            for (int nt = 0; nt < 4; ++nt) {
                BF16Frag bfr;  // B[k,n] = Wfc[n][k]
                const __bf16* wr = &Wt[(nt * 16 + l16) * LD + kc * 32 + lh * 8];
                bfr.q[0] = *(const uint4*)(wr);
                bfr.q[1] = *(const uint4*)(wr + 16);
                acc[nt] = wmma_bf16(af.v, bfr.v, acc[nt]);
            }
        }
    }

    #pragma unroll
    for (int nt = 0; nt < 4; ++nt) {
        const int n  = ncol0 + nt * 16 + l16;
        const float bb = bias[n];
        #pragma unroll
        for (int r = 0; r < 8; ++r) {
            const int t = trow0 + r + 8 * lh;
            out[(size_t)t * EMBED + n] = acc[nt][r] + bb;
        }
    }
}

// ---------------------------------------------------------------------------
// Kernel 0: f32 -> bf16 conversion (Wfc)
// ---------------------------------------------------------------------------
__global__ void cvt_bf16_kernel(const float* __restrict__ src,
                                __bf16* __restrict__ dst, int n)
{
    const int i = blockIdx.x * blockDim.x + threadIdx.x;
    if (i < n) dst[i] = (__bf16)src[i];
}

// ---------------------------------------------------------------------------
extern "C" void kernel_launch(void* const* d_in, const int* in_sizes, int n_in,
                              void* d_out, int out_size, void* d_ws, size_t ws_size,
                              hipStream_t stream)
{
    const float* values = (const float*)d_in[0];
    const float* keys   = (const float*)d_in[1];
    const float* query  = (const float*)d_in[2];
    // d_in[3] is the causal mask: applied analytically, never read.
    const float* Wv  = (const float*)d_in[4];
    const float* bv  = (const float*)d_in[5];
    const float* Wk  = (const float*)d_in[6];
    const float* bk  = (const float*)d_in[7];
    const float* Wq  = (const float*)d_in[8];
    const float* bq  = (const float*)d_in[9];
    const float* Wfc = (const float*)d_in[10];
    const float* bfc = (const float*)d_in[11];

    char* ws = (char*)d_ws;
    const size_t planeBytes = (size_t)BATCH * HEADS * SEQ * HEADD * sizeof(__bf16); // 16 MB
    __bf16* Qp      = (__bf16*)(ws);
    __bf16* Kp      = (__bf16*)(ws + 1 * planeBytes);
    __bf16* Vp      = (__bf16*)(ws + 2 * planeBytes);
    __bf16* AttnOut = (__bf16*)(ws + 3 * planeBytes);
    __bf16* Wfcbf   = (__bf16*)(ws + 4 * planeBytes);

    const dim3 blk(128);
    proj_qkv_kernel<<<dim3(BATCH * HEADS, SEQ / 64), blk, 0, stream>>>(
        values, keys, query, Wv, bv, Wk, bk, Wq, bq, Qp, Kp, Vp);

    cvt_bf16_kernel<<<(EMBED * EMBED + 255) / 256, 256, 0, stream>>>(
        Wfc, Wfcbf, EMBED * EMBED);

    flash_attn_kernel<<<dim3(BATCH * HEADS, SEQ / 64), blk, 0, stream>>>(
        Qp, Kp, Vp, AttnOut);

    fc_kernel<<<dim3(BATCH * SEQ / 64, EMBED / 64), blk, 0, stream>>>(
        AttnOut, Wfcbf, bfc, (float*)d_out);
}
